// DeepFM_41334765257176
// MI455X (gfx1250) — matrix operand order repaired and checked by reading, not compiled, not run on hardware
//
#include <hip/hip_runtime.h>
#include <hip/hip_bf16.h>

// ---------------- types for WMMA ----------------
typedef __attribute__((ext_vector_type(16))) __bf16 v16bf;
typedef __attribute__((ext_vector_type(8)))  __bf16 v8bf;
typedef __attribute__((ext_vector_type(8)))  float  v8f;

union Frag {
    v16bf f;
    v8bf  h[2];
};

// ---------------- problem constants ----------------
#define NN   16384
#define FF   39
#define VV   50000
#define EE   64
#define K1   2496   // F*E
#define H1   1024
#define H2   512
#define H3   256
#define BN_EPS 1e-5f

// =====================================================================
// 1) Embedding gather + first-order + FM second-order partial per row.
//    One block per sample, 64 threads (= E).
// =====================================================================
__global__ __launch_bounds__(64) void embed_kernel(
    const long long* __restrict__ Xi, const float* __restrict__ Xv,
    const float* __restrict__ W1, const float* __restrict__ W2,
    const float* __restrict__ bias,
    __bf16* __restrict__ deep, float* __restrict__ p)
{
    __shared__ float red[64];
    const int n = blockIdx.x;
    const int t = threadIdx.x;

    float s = 0.f, q = 0.f;
    for (int f = 0; f < FF; ++f) {
        const int   idx = (int)Xi[(size_t)n * FF + f];
        const float xv  = Xv[(size_t)n * FF + f];
        const float w   = W2[((size_t)f * VV + idx) * EE + t] * xv;
        deep[(size_t)n * K1 + f * EE + t] = (__bf16)w;
        s += w;
        q += w * w;
    }
    float val = 0.5f * (s * s - q);          // fm2 component t
    if (t < FF) {                            // first-order component t
        const int idx = (int)Xi[(size_t)n * FF + t];
        val += W1[(size_t)t * VV + idx] * Xv[(size_t)n * FF + t];
    }
    red[t] = val;
    __syncthreads();
    for (int st = 32; st > 0; st >>= 1) {
        if (t < st) red[t] += red[t + st];
        __syncthreads();
    }
    if (t == 0) p[n] = red[0] + bias[0];
}

// =====================================================================
// 2) Convert fp32 weights (K x N, row major) -> bf16 transposed (N x K)
// =====================================================================
__global__ void wconv_kernel(const float* __restrict__ W, __bf16* __restrict__ Wt,
                             int K, int N)
{
    size_t i = (size_t)blockIdx.x * blockDim.x + threadIdx.x;
    size_t total = (size_t)K * N;
    if (i < total) {
        int nn = (int)(i / K);
        int kk = (int)(i % K);
        Wt[i] = (__bf16)W[(size_t)kk * N + nn];
    }
}

// =====================================================================
// 3) bf16 WMMA GEMM: Z(MxN) = A(MxK) * Bt(NxK)^T + lb, f32 accumulate.
//    256 threads = 8 waves; wave tile 64x32 (4x2 WMMA tiles); block 128x128.
//    A fragment layout per ISA 16-bit A 16x32: lane = M%16 row,
//      kb=(lane/16)*8, VGPR0-3: K=kb..kb+7, VGPR4-7: K=kb+16..kb+23.
//    B fragment 32x16, same K order as A: lane = N col, kb=(lane/16)*8,
//      VGPR0-3: K=kb..kb+7, VGPR4-7: K=kb+16..kb+23 (same as A).
// =====================================================================
__global__ __launch_bounds__(256) void gemm_bf16_kernel(
    const __bf16* __restrict__ A, const __bf16* __restrict__ Bt,
    const float* __restrict__ lb, float* __restrict__ Z,
    int M, int N, int K)
{
    const int lane  = threadIdx.x & 31;
    const int wave  = threadIdx.x >> 5;
    const int half  = lane >> 4;     // 0 | 1
    const int l16   = lane & 15;

    const int nBlk  = N >> 7;        // N / 128
    const int bm    = blockIdx.x / nBlk;
    const int bn    = blockIdx.x % nBlk;
    const int m0    = bm * 128 + (wave >> 2) * 64;
    const int n0    = bn * 128 + (wave & 3) * 32;

    v8f acc[4][2];
#pragma unroll
    for (int i = 0; i < 4; ++i)
#pragma unroll
        for (int j = 0; j < 2; ++j) acc[i][j] = (v8f){};

    const __bf16* arow[4];
#pragma unroll
    for (int mt = 0; mt < 4; ++mt)
        arow[mt] = A + (size_t)(m0 + mt * 16 + l16) * K;
    const __bf16* brow[2];
#pragma unroll
    for (int nt = 0; nt < 2; ++nt)
        brow[nt] = Bt + (size_t)(n0 + nt * 16 + l16) * K;

    const int kbA = half * 8;    // A lane K base
    const int kbB = half * 8;    // B lane K base (same as A)

    for (int k0 = 0; k0 < K; k0 += 32) {
        Frag a[4], b[2];
#pragma unroll
        for (int mt = 0; mt < 4; ++mt) {
            a[mt].h[0] = *reinterpret_cast<const v8bf*>(arow[mt] + k0 + kbA);
            a[mt].h[1] = *reinterpret_cast<const v8bf*>(arow[mt] + k0 + kbA + 16);
        }
#pragma unroll
        for (int nt = 0; nt < 2; ++nt) {
            b[nt].h[0] = *reinterpret_cast<const v8bf*>(brow[nt] + k0 + kbB);
            b[nt].h[1] = *reinterpret_cast<const v8bf*>(brow[nt] + k0 + kbB + 16);
        }
#pragma unroll
        for (int mt = 0; mt < 4; ++mt)
#pragma unroll
            for (int nt = 0; nt < 2; ++nt)
                acc[mt][nt] = __builtin_amdgcn_wmma_f32_16x16x32_bf16(
                    false, a[mt].f, false, b[nt].f,
                    (short)0, acc[mt][nt], false, false);
    }

    // epilogue: C/D layout — VGPR r: lanes0-15 (M=r, N=l16), lanes16-31 (M=r+8)
#pragma unroll
    for (int mt = 0; mt < 4; ++mt) {
#pragma unroll
        for (int nt = 0; nt < 2; ++nt) {
            const int col   = n0 + nt * 16 + l16;
            const float bv  = lb[col];
            const int rbase = m0 + mt * 16 + half * 8;
#pragma unroll
            for (int r = 0; r < 8; ++r)
                Z[(size_t)(rbase + r) * N + col] = acc[mt][nt][r] + bv;
        }
    }
}

// =====================================================================
// 4) BatchNorm statistics (column mean/var over N rows)
// =====================================================================
__global__ void zero2_kernel(float* a, float* b, int H)
{
    int j = blockIdx.x * blockDim.x + threadIdx.x;
    if (j < H) { a[j] = 0.f; b[j] = 0.f; }
}

__global__ __launch_bounds__(256) void bn_partial_kernel(
    const float* __restrict__ Z, float* __restrict__ sum,
    float* __restrict__ sumsq, int H, int rowsPerChunk)
{
    const int j  = blockIdx.x * blockDim.x + threadIdx.x;  // column
    const int r0 = blockIdx.y * rowsPerChunk;
    float s = 0.f, q = 0.f;
    for (int r = 0; r < rowsPerChunk; ++r) {
        float v = Z[(size_t)(r0 + r) * H + j];
        s += v;
        q += v * v;
    }
    atomicAdd(&sum[j], s);
    atomicAdd(&sumsq[j], q);
}

__global__ void bn_finalize_kernel(const float* __restrict__ sum,
                                   const float* __restrict__ sumsq,
                                   const float* __restrict__ g,
                                   const float* __restrict__ b,
                                   float* __restrict__ scale,
                                   float* __restrict__ shift, int H)
{
    int j = blockIdx.x * blockDim.x + threadIdx.x;
    if (j < H) {
        const float invN = 1.f / (float)NN;
        float m   = sum[j] * invN;
        float var = sumsq[j] * invN - m * m;
        float sc  = g[j] * rsqrtf(var + BN_EPS);
        scale[j] = sc;
        shift[j] = b[j] - m * sc;
    }
}

// BN + ReLU + convert to bf16 for the next GEMM
__global__ void bn_apply_kernel(const float* __restrict__ Z,
                                const float* __restrict__ scale,
                                const float* __restrict__ shift,
                                __bf16* __restrict__ Hout, int H, size_t total)
{
    size_t i = (size_t)blockIdx.x * blockDim.x + threadIdx.x;
    if (i < total) {
        int j = (int)(i % H);
        float v = Z[i] * scale[j] + shift[j];
        Hout[i] = (__bf16)(v > 0.f ? v : 0.f);
    }
}

// =====================================================================
// 5) Final: out[n] = p[n] + sum_j relu(bn(z3[n,j]))
// =====================================================================
__global__ __launch_bounds__(256) void final_kernel(
    const float* __restrict__ Z3, const float* __restrict__ scale,
    const float* __restrict__ shift, const float* __restrict__ p,
    float* __restrict__ out)
{
    __shared__ float red[256];
    const int n = blockIdx.x;
    const int t = threadIdx.x;
    float v = Z3[(size_t)n * H3 + t] * scale[t] + shift[t];
    red[t] = v > 0.f ? v : 0.f;
    __syncthreads();
    for (int s = 128; s > 0; s >>= 1) {
        if (t < s) red[t] += red[t + s];
        __syncthreads();
    }
    if (t == 0) out[n] = red[0] + p[n];
}

// =====================================================================
// Host launcher
// =====================================================================
extern "C" void kernel_launch(void* const* d_in, const int* in_sizes, int n_in,
                              void* d_out, int out_size, void* d_ws, size_t ws_size,
                              hipStream_t stream)
{
    (void)in_sizes; (void)n_in; (void)out_size; (void)ws_size;

    const long long* Xi  = (const long long*)d_in[0];
    const float* Xv   = (const float*)d_in[1];
    const float* W1   = (const float*)d_in[2];
    const float* W2   = (const float*)d_in[3];
    const float* bias = (const float*)d_in[4];
    const float* lw1  = (const float*)d_in[5];
    const float* lb1  = (const float*)d_in[6];
    const float* g1   = (const float*)d_in[7];
    const float* b1   = (const float*)d_in[8];
    const float* lw2  = (const float*)d_in[9];
    const float* lb2  = (const float*)d_in[10];
    const float* g2   = (const float*)d_in[11];
    const float* b2   = (const float*)d_in[12];
    const float* lw3  = (const float*)d_in[13];
    const float* lb3  = (const float*)d_in[14];
    const float* g3   = (const float*)d_in[15];
    const float* b3   = (const float*)d_in[16];
    float* out = (float*)d_out;

    // -------- workspace layout (all sizes multiples of 256 bytes) --------
    char* ws = (char*)d_ws;
    const size_t sz_deep = (size_t)NN * K1 * 2;   // 81,788,928
    const size_t sz_wt1  = (size_t)K1 * H1 * 2;   // 5,111,808
    const size_t sz_wt2  = (size_t)H1 * H2 * 2;   // 1,048,576
    const size_t sz_wt3  = (size_t)H2 * H3 * 2;   //   262,144
    const size_t sz_z1   = (size_t)NN * H1 * 4;   // 67,108,864
    const size_t sz_h1   = (size_t)NN * H1 * 2;   // 33,554,432

    size_t off = 0;
    char* deep = ws + off; off += sz_deep;
    char* wt1  = ws + off; off += sz_wt1;
    char* wt2  = ws + off; off += sz_wt2;
    char* wt3  = ws + off; off += sz_wt3;
    char* z1   = ws + off; off += sz_z1;
    char* h1   = ws + off; off += sz_h1;
    char* pbuf = ws + off; off += (size_t)NN * 4;
    char* ssum = ws + off; off += 4096;
    char* ssq  = ws + off; off += 4096;
    char* ssc  = ws + off; off += 4096;
    char* ssh  = ws + off; off += 4096;
    // reuse the deep region once it is consumed by GEMM1:
    char* z2 = deep;                              // NN*H2*4 = 33.5 MB
    char* h2 = deep + (size_t)NN * H2 * 4;        // NN*H2*2 = 16.8 MB
    char* z3 = h2   + (size_t)NN * H2 * 2;        // NN*H3*4 = 16.8 MB

    // -------- 1) embeddings + FM partials --------
    embed_kernel<<<NN, 64, 0, stream>>>(Xi, Xv, W1, W2, bias,
                                        (__bf16*)deep, (float*)pbuf);

    // -------- 2) weight transpose+convert --------
    {
        size_t t1 = (size_t)K1 * H1, t2 = (size_t)H1 * H2, t3 = (size_t)H2 * H3;
        wconv_kernel<<<(unsigned)((t1 + 255) / 256), 256, 0, stream>>>(lw1, (__bf16*)wt1, K1, H1);
        wconv_kernel<<<(unsigned)((t2 + 255) / 256), 256, 0, stream>>>(lw2, (__bf16*)wt2, H1, H2);
        wconv_kernel<<<(unsigned)((t3 + 255) / 256), 256, 0, stream>>>(lw3, (__bf16*)wt3, H2, H3);
    }

    // -------- layer 1 --------
    gemm_bf16_kernel<<<(NN / 128) * (H1 / 128), 256, 0, stream>>>(
        (const __bf16*)deep, (const __bf16*)wt1, lb1, (float*)z1, NN, H1, K1);
    zero2_kernel<<<(H1 + 255) / 256, 256, 0, stream>>>((float*)ssum, (float*)ssq, H1);
    bn_partial_kernel<<<dim3(H1 / 256, NN / 256), 256, 0, stream>>>(
        (const float*)z1, (float*)ssum, (float*)ssq, H1, 256);
    bn_finalize_kernel<<<(H1 + 255) / 256, 256, 0, stream>>>(
        (const float*)ssum, (const float*)ssq, g1, b1, (float*)ssc, (float*)ssh, H1);
    bn_apply_kernel<<<(unsigned)(((size_t)NN * H1 + 255) / 256), 256, 0, stream>>>(
        (const float*)z1, (const float*)ssc, (const float*)ssh, (__bf16*)h1, H1, (size_t)NN * H1);

    // -------- layer 2 --------
    gemm_bf16_kernel<<<(NN / 128) * (H2 / 128), 256, 0, stream>>>(
        (const __bf16*)h1, (const __bf16*)wt2, lb2, (float*)z2, NN, H2, H1);
    zero2_kernel<<<(H2 + 255) / 256, 256, 0, stream>>>((float*)ssum, (float*)ssq, H2);
    bn_partial_kernel<<<dim3(H2 / 256, NN / 256), 256, 0, stream>>>(
        (const float*)z2, (float*)ssum, (float*)ssq, H2, 256);
    bn_finalize_kernel<<<(H2 + 255) / 256, 256, 0, stream>>>(
        (const float*)ssum, (const float*)ssq, g2, b2, (float*)ssc, (float*)ssh, H2);
    bn_apply_kernel<<<(unsigned)(((size_t)NN * H2 + 255) / 256), 256, 0, stream>>>(
        (const float*)z2, (const float*)ssc, (const float*)ssh, (__bf16*)h2, H2, (size_t)NN * H2);

    // -------- layer 3 --------
    gemm_bf16_kernel<<<(NN / 128) * (H3 / 128), 256, 0, stream>>>(
        (const __bf16*)h2, (const __bf16*)wt3, lb3, (float*)z3, NN, H3, H2);
    zero2_kernel<<<(H3 + 255) / 256, 256, 0, stream>>>((float*)ssum, (float*)ssq, H3);
    bn_partial_kernel<<<dim3(H3 / 256, NN / 256), 256, 0, stream>>>(
        (const float*)z3, (float*)ssum, (float*)ssq, H3, 256);
    bn_finalize_kernel<<<(H3 + 255) / 256, 256, 0, stream>>>(
        (const float*)ssum, (const float*)ssq, g3, b3, (float*)ssc, (float*)ssh, H3);

    // -------- final reduction --------
    final_kernel<<<NN, 256, 0, stream>>>((const float*)z3, (const float*)ssc,
                                         (const float*)ssh, (const float*)pbuf, out);
}
